// HierarchUpdateMlp_24223615550197
// MI455X (gfx1250) — matrix operand order repaired and checked by reading, not compiled, not run on hardware
//
#include <hip/hip_runtime.h>
#include <hip/hip_bf16.h>

// ---------------------------------------------------------------------------
// HierarchUpdateMlp on MI455X (gfx1250): bf16 WMMA pipeline, f32 accumulate.
//   update [N,17,256] f32  ->  out [N,256] f32
// Transposed chain: every layer computes Y^T = W * X^T, so weights are the
// WMMA A operand and activations the B operand (lane = sample). D output
// (lane = sample) feeds the next layer's B operand directly via one packed
// ds_store_b128 per column tile. One wave32 per 16-sample tile; layer2+final
// fused; 32 KB LDS -> 10 WG/WGP. __launch_bounds__(32,3) caps VGPRs so the
// allocator cannot balloon past the LDS-limited occupancy. Weights are
// pre-converted to bf16 in d_ws (L2-resident, ~1.9 MB).
// ---------------------------------------------------------------------------

typedef __attribute__((ext_vector_type(16))) __bf16 v16bf;
typedef __attribute__((ext_vector_type(2)))  __bf16 v2bf;
typedef __attribute__((ext_vector_type(8)))  float  v8f;

union Frag16 { unsigned int u[8]; v16bf v; };

// Exact RNE scalar convert (weight prep only; not perf-critical).
__device__ __forceinline__ unsigned short f2bf(float f) {
    unsigned int x = __float_as_uint(f);
    x += 0x7FFFu + ((x >> 16) & 1u);
    return (unsigned short)(x >> 16);
}

// Fast packed f32x2 -> bf16x2.
__device__ __forceinline__ unsigned int pack2(float lo, float hi) {
#if __has_builtin(__builtin_amdgcn_cvt_pk_bf16_f32)
    v2bf r = __builtin_amdgcn_cvt_pk_bf16_f32(lo, hi);
    return __builtin_bit_cast(unsigned int, r);
#else
    // round-half-up + byte-perm pack: 3 VALU ops per 2 elements
    unsigned int a = __float_as_uint(lo) + 0x8000u;
    unsigned int b = __float_as_uint(hi) + 0x8000u;
    return __builtin_amdgcn_perm(b, a, 0x07060302u);  // {b.hi16, a.hi16}
#endif
}

// Weight fragment = WMMA A operand (16 neurons x 32 K), row-major W[n][k].
// A k order used here (contiguous, matches B; not AMD's table order): lane L -> M = L%16; VGPR v -> K = (L/16)*16 + v*2.
__device__ __forceinline__ v16bf load_wfrag(const unsigned short* __restrict__ p, int laneHi) {
    const uint4 q0 = *(const uint4*)(p + laneHi * 16);
    const uint4 q1 = *(const uint4*)(p + laneHi * 16 + 8);
    Frag16 A;
    A.u[0] = q0.x; A.u[1] = q0.y; A.u[2] = q0.z; A.u[3] = q0.w;
    A.u[4] = q1.x; A.u[5] = q1.y; A.u[6] = q1.z; A.u[7] = q1.w;
    return A.v;
}

// Activation fragment = WMMA B operand (32 K x 16 samples) from f32 memory.
// B k order used here (contiguous, matches A; not AMD's table order): lanes 0-15 -> K=0..15, lanes 16-31 -> K=16..31; 2 K per VGPR.
// Per lane: 16 consecutive floats of its sample row starting at K = laneHi*16.
__device__ __forceinline__ v16bf load_xfrag_f32(const float* __restrict__ p, int laneHi) {
    const float4 f0 = *(const float4*)(p + laneHi * 16);
    const float4 f1 = *(const float4*)(p + laneHi * 16 + 4);
    const float4 f2 = *(const float4*)(p + laneHi * 16 + 8);
    const float4 f3 = *(const float4*)(p + laneHi * 16 + 12);
    Frag16 B;
    B.u[0] = pack2(f0.x, f0.y); B.u[1] = pack2(f0.z, f0.w);
    B.u[2] = pack2(f1.x, f1.y); B.u[3] = pack2(f1.z, f1.w);
    B.u[4] = pack2(f2.x, f2.y); B.u[5] = pack2(f2.z, f2.w);
    B.u[6] = pack2(f3.x, f3.y); B.u[7] = pack2(f3.z, f3.w);
    return B.v;
}

// Activation fragment from dword-packed bf16 LDS tile: 2x ds_load_b128.
// p points at dword (sampleRow*64 + kWindow*16); lane adds laneHi*8 dwords.
__device__ __forceinline__ v16bf load_xfrag_lds(const unsigned int* p, int laneHi) {
    const uint4 q0 = *(const uint4*)(p + laneHi * 8);
    const uint4 q1 = *(const uint4*)(p + laneHi * 8 + 4);
    Frag16 B;
    B.u[0] = q0.x; B.u[1] = q0.y; B.u[2] = q0.z; B.u[3] = q0.w;
    B.u[4] = q1.x; B.u[5] = q1.y; B.u[6] = q1.z; B.u[7] = q1.w;
    return B.v;
}

// Epilogue: bias (per-neuron = per-reg) + ReLU + pack to bf16 dwords,
// one ds_store_b128. dst = &ldsDwords[sampleRow*64 + ct*8 + laneHi*4].
__device__ __forceinline__ void store_pack_lds(unsigned int* dst, const v8f acc,
                                               const float* __restrict__ bias) {
    const float4 bb0 = *(const float4*)(bias);
    const float4 bb1 = *(const float4*)(bias + 4);
    uint4 q;
    q.x = pack2(fmaxf(acc[0] + bb0.x, 0.f), fmaxf(acc[1] + bb0.y, 0.f));
    q.y = pack2(fmaxf(acc[2] + bb0.z, 0.f), fmaxf(acc[3] + bb0.w, 0.f));
    q.z = pack2(fmaxf(acc[4] + bb1.x, 0.f), fmaxf(acc[5] + bb1.y, 0.f));
    q.w = pack2(fmaxf(acc[6] + bb1.z, 0.f), fmaxf(acc[7] + bb1.w, 0.f));
    *(uint4*)dst = q;
}

#define WMMA_BF16(A, B, C) \
    __builtin_amdgcn_wmma_f32_16x16x32_bf16(false, (A), false, (B), (short)0, (C), false, false)

// f32->bf16 weight conversion (runs once per launch; weights stay in L2).
__global__ void hmlp_cvt_bf16(const float* __restrict__ s, unsigned short* __restrict__ d, int n) {
    int i = blockIdx.x * blockDim.x + threadIdx.x;
    if (i < n) d[i] = f2bf(s[i]);
}

__launch_bounds__(32, 3)
__global__ void hmlp_wmma_kernel(
    const float* __restrict__ update,            // [N,17,256]
    const float* __restrict__ b10,               // [128]
    const float* __restrict__ b1,                // [6,128]
    const float* __restrict__ b2,                // [6,128]
    const float* __restrict__ bfv,               // [256]
    const unsigned short* __restrict__ w10,      // bf16 [128,1280]
    const unsigned short* __restrict__ w1,       // bf16 [6,128,512]
    const unsigned short* __restrict__ w2,       // bf16 [6,128,256]
    const unsigned short* __restrict__ wf,       // bf16 [256,768]
    float* __restrict__ out,                     // [N,256]
    int N)
{
    // out1^T staged as dword-packed bf16: [group 0..6][sample 0..15][64 dwords]
    __shared__ __align__(16) unsigned int out1d[7 * 16 * 64];  // 28 KB
    __shared__ __align__(16) unsigned int s2d[16 * 64];        //  4 KB

    const int lane   = threadIdx.x & 31;
    const int laneLo = lane & 15;
    const int laneHi = lane >> 4;
    const int row0   = blockIdx.x * 16;
    const int row    = row0 + laneLo;
    const int rowA   = (row < N) ? row : (N - 1);   // clamp for loads
    const float* __restrict__ arow = update + (size_t)rowA * (17 * 256);

    // Warm the WGP cache with the pair joints while group-0 computes.
#pragma unroll
    for (int j = 5; j < 17; ++j)
        __builtin_prefetch(arow + j * 256, 0, 3);

    // ---------------- Layer 1, group 0: W1_0 x x0^T -> out1d[0] ------------
    {
        v8f acc[8];
#pragma unroll
        for (int c = 0; c < 8; ++c) {
#pragma unroll
            for (int r = 0; r < 8; ++r) acc[c][r] = 0.f;
        }
#pragma unroll 1
        for (int ks = 0; ks < 40; ++ks) {                 // K = 1280
            v16bf X = load_xfrag_f32(arow + ks * 32, laneHi);
#pragma unroll
            for (int c = 0; c < 8; ++c) {
                v16bf W = load_wfrag(w10 + (size_t)(c * 16 + laneLo) * 1280 + ks * 32, laneHi);
                acc[c] = WMMA_BF16(W, X, acc[c]);
            }
        }
#pragma unroll
        for (int c = 0; c < 8; ++c)
            store_pack_lds(&out1d[(0 * 16 + laneLo) * 64 + c * 8 + laneHi * 4],
                           acc[c], b10 + c * 16 + laneHi * 8);
    }

    // ---------------- Layer 1, pair groups 1..6 ----------------------------
    {
        const int P1a[6] = {5, 7, 8, 11, 13, 14};
        const int P1b[6] = {6, 9, 10, 12, 15, 16};
#pragma unroll
        for (int g = 0; g < 6; ++g) {
            v8f acc[8];
#pragma unroll
            for (int c = 0; c < 8; ++c) {
#pragma unroll
                for (int r = 0; r < 8; ++r) acc[c][r] = 0.f;
            }
#pragma unroll 1
            for (int ks = 0; ks < 16; ++ks) {             // K = 512 = 2 joints
                int joint = (ks < 8) ? P1a[g] : P1b[g];
                v16bf X = load_xfrag_f32(arow + joint * 256 + (ks & 7) * 32, laneHi);
#pragma unroll
                for (int c = 0; c < 8; ++c) {
                    v16bf W = load_wfrag(w1 + (size_t)(g * 128 + c * 16 + laneLo) * 512 + ks * 32, laneHi);
                    acc[c] = WMMA_BF16(W, X, acc[c]);
                }
            }
#pragma unroll
            for (int c = 0; c < 8; ++c)
                store_pack_lds(&out1d[((g + 1) * 16 + laneLo) * 64 + c * 8 + laneHi * 4],
                               acc[c], b1 + g * 128 + c * 16 + laneHi * 8);
        }
    }
    __syncthreads();   // single wave: cheap; orders LDS writes vs reads

    // -------- Fused layer 2 + final layer ----------------------------------
    // For each 128-wide output half: recompute layer-2 per group into the
    // 4 KB scratch tile, then immediately accumulate its K-slice of the final
    // GEMM (removes the 24 KB out2 buffer; occupancy: 10 WG/WGP).
    const int P2a[6] = {0, 1, 1, 1, 4, 4};
    const int P2b[6] = {1, 2, 3, 4, 5, 6};
#pragma unroll 1
    for (int h = 0; h < 2; ++h) {
        v8f facc[8];
#pragma unroll
        for (int c = 0; c < 8; ++c) {
#pragma unroll
            for (int r = 0; r < 8; ++r) facc[c][r] = 0.f;
        }
#pragma unroll
        for (int g = 0; g < 6; ++g) {
            // ---- layer-2 group g: W2[g] x pair(out1)^T -> acc ----
            v8f acc[8];
#pragma unroll
            for (int c = 0; c < 8; ++c) {
#pragma unroll
                for (int r = 0; r < 8; ++r) acc[c][r] = 0.f;
            }
#pragma unroll 1
            for (int ks = 0; ks < 8; ++ks) {              // K = 256 = 2 segs
                int src = (ks < 4) ? P2a[g] : P2b[g];
                v16bf X = load_xfrag_lds(&out1d[(src * 16 + laneLo) * 64 + (ks & 3) * 16], laneHi);
#pragma unroll
                for (int c = 0; c < 8; ++c) {
                    v16bf W = load_wfrag(w2 + (size_t)(g * 128 + c * 16 + laneLo) * 256 + ks * 32, laneHi);
                    acc[c] = WMMA_BF16(W, X, acc[c]);
                }
            }
            // ---- bias+relu -> packed bf16 scratch tile ----
#pragma unroll
            for (int c = 0; c < 8; ++c)
                store_pack_lds(&s2d[laneLo * 64 + c * 8 + laneHi * 4],
                               acc[c], b2 + g * 128 + c * 16 + laneHi * 8);
            __syncthreads();
            // ---- final GEMM K-slice g*128..g*128+127 into facc ----
#pragma unroll 1
            for (int k = 0; k < 4; ++k) {
                v16bf X = load_xfrag_lds(&s2d[laneLo * 64 + k * 16], laneHi);
#pragma unroll
                for (int c = 0; c < 8; ++c) {
                    v16bf W = load_wfrag(wf + (size_t)(h * 128 + c * 16 + laneLo) * 768 + g * 128 + k * 32, laneHi);
                    facc[c] = WMMA_BF16(W, X, facc[c]);
                }
            }
            __syncthreads();   // s2d reused by next group
        }
        // ---- write this 128-wide half: lane owns one sample row ----
        const int rr = row0 + laneLo;
        if (rr < N) {
#pragma unroll
            for (int c = 0; c < 8; ++c) {
                const float* bp = bfv + h * 128 + c * 16 + laneHi * 8;
                const float4 bb0 = *(const float4*)(bp);
                const float4 bb1 = *(const float4*)(bp + 4);
                float4 o0, o1;
                o0.x = fmaxf(facc[c][0] + bb0.x, 0.f);
                o0.y = fmaxf(facc[c][1] + bb0.y, 0.f);
                o0.z = fmaxf(facc[c][2] + bb0.z, 0.f);
                o0.w = fmaxf(facc[c][3] + bb0.w, 0.f);
                o1.x = fmaxf(facc[c][4] + bb1.x, 0.f);
                o1.y = fmaxf(facc[c][5] + bb1.y, 0.f);
                o1.z = fmaxf(facc[c][6] + bb1.z, 0.f);
                o1.w = fmaxf(facc[c][7] + bb1.w, 0.f);
                float* po = out + (size_t)rr * 256 + h * 128 + c * 16 + laneHi * 8;
                *(float4*)(po)     = o0;
                *(float4*)(po + 4) = o1;
            }
        }
    }
}

extern "C" void kernel_launch(void* const* d_in, const int* in_sizes, int n_in,
                              void* d_out, int out_size, void* d_ws, size_t ws_size,
                              hipStream_t stream) {
    const float* update = (const float*)d_in[0];   // [N,17,256]
    const float* W10f   = (const float*)d_in[1];   // [128,1280]
    const float* b10    = (const float*)d_in[2];   // [128]
    const float* W1f    = (const float*)d_in[3];   // [6,128,512]
    const float* b1     = (const float*)d_in[4];   // [6,128]
    const float* W2f    = (const float*)d_in[5];   // [6,128,256]
    const float* b2     = (const float*)d_in[6];   // [6,128]
    const float* Wff    = (const float*)d_in[7];   // [256,768]
    const float* bfv    = (const float*)d_in[8];   // [256]
    float* out          = (float*)d_out;

    const int N = in_sizes[0] / (17 * 256);

    // bf16 weight staging in workspace (1.9 MB total; L2-resident).
    const int n10 = 128 * 1280;       // 163840
    const int n1  = 6 * 128 * 512;    // 393216
    const int n2  = 6 * 128 * 256;    // 196608
    const int nf  = 256 * 768;        // 196608
    unsigned short* w10 = (unsigned short*)d_ws;
    unsigned short* w1  = w10 + n10;
    unsigned short* w2  = w1 + n1;
    unsigned short* wf  = w2 + n2;

    hmlp_cvt_bf16<<<(n10 + 255) / 256, 256, 0, stream>>>(W10f, w10, n10);
    hmlp_cvt_bf16<<<(n1  + 255) / 256, 256, 0, stream>>>(W1f,  w1,  n1);
    hmlp_cvt_bf16<<<(n2  + 255) / 256, 256, 0, stream>>>(W2f,  w2,  n2);
    hmlp_cvt_bf16<<<(nf  + 255) / 256, 256, 0, stream>>>(Wff,  wf,  nf);

    const int tiles = (N + 15) / 16;
    hmlp_wmma_kernel<<<tiles, 32, 0, stream>>>(update, b10, b1, b2, bfv,
                                               w10, w1, w2, wf, out, N);
}
